// FAR_TransformerBlock_40492951667581
// MI455X (gfx1250) — matrix operand order repaired and checked by reading, not compiled, not run on hardware
//
#include <hip/hip_runtime.h>

// ---------------------------------------------------------------------------
// Types for CDNA5 WMMA (wave32, 16x16x32 bf16 -> f32)
// ---------------------------------------------------------------------------
typedef __bf16 bf16;
typedef __attribute__((ext_vector_type(8)))  __bf16 bf16x8;
typedef __attribute__((ext_vector_type(16))) __bf16 bf16x16;
typedef __attribute__((ext_vector_type(8)))  float  f32x8;
typedef __attribute__((ext_vector_type(4)))  int    i32x4;

#define AS1 __attribute__((address_space(1)))
#define AS3 __attribute__((address_space(3)))

// Async global->LDS copy path (CDNA5 GLOBAL_LOAD_ASYNC_TO_LDS_B128, ASYNCcnt)
#if defined(__has_builtin)
#if __has_builtin(__builtin_amdgcn_global_load_async_to_lds_b128)
#define USE_ASYNC_LDS 1
#else
#define USE_ASYNC_LDS 0
#endif
#else
#define USE_ASYNC_LDS 0
#endif

#if USE_ASYNC_LDS
#define TILE_COPY(dstLds, srcGlb)                                      \
  __builtin_amdgcn_global_load_async_to_lds_b128(                      \
      (AS1 i32x4*)(srcGlb), (AS3 i32x4*)(dstLds), 0, 0)
#if __has_builtin(__builtin_amdgcn_s_wait_asynccnt)
#define WAIT_ASYNC(n) __builtin_amdgcn_s_wait_asynccnt(n)
#else
#define WAIT_ASYNC(n) asm volatile("s_wait_asynccnt %0" ::"n"(n) : "memory")
#endif
#else
#define TILE_COPY(dstLds, srcGlb) \
  *(bf16x8*)(dstLds) = *(const bf16x8*)(srcGlb)
#define WAIT_ASYNC(n)
#endif

__device__ __forceinline__ f32x8 wmma_bf16(bf16x16 a, bf16x16 b, f32x8 c) {
  return __builtin_amdgcn_wmma_f32_16x16x32_bf16(false, a, false, b, (short)0, c,
                                                 false, false);
}

__device__ __forceinline__ bf16x16 cat8(bf16x8 lo, bf16x8 hi) {
  return __builtin_shufflevector(lo, hi, 0, 1, 2, 3, 4, 5, 6, 7, 8, 9, 10, 11,
                                 12, 13, 14, 15);
}

__device__ __forceinline__ float gelu_tanh(float x) {
  float x3 = x * x * x;
  return 0.5f * x * (1.0f + tanhf(0.7978845608028654f * (x + 0.044715f * x3)));
}

// ---------------------------------------------------------------------------
// fp32 (K x N) -> bf16 transposed (N x K); K is a power of two (kshift)
// ---------------------------------------------------------------------------
__global__ void conv_wT(const float* __restrict__ W, bf16* __restrict__ Wt,
                        int kshift, int N) {
  size_t i = (size_t)blockIdx.x * 256 + threadIdx.x;
  size_t K = (size_t)1 << kshift;
  if (i >= K * N) return;
  size_t n = i >> kshift;
  size_t k = i & (K - 1);
  Wt[i] = (bf16)W[k * (size_t)N + n];
}

// ---------------------------------------------------------------------------
// e = silu(temb) @ W + b        (M = B rows, tiny K=D dot per output column)
// ---------------------------------------------------------------------------
__global__ __launch_bounds__(256) void temb_gemm(
    const float* __restrict__ temb, const float* __restrict__ W,
    const float* __restrict__ bias, float* __restrict__ E, int D, int N3) {
  __shared__ float st[2048];
  int b = blockIdx.y;
  int n = blockIdx.x * 256 + threadIdx.x;
  for (int i = threadIdx.x; i < D; i += 256) {
    float v = temb[(size_t)b * D + i];
    st[i] = v / (1.0f + __expf(-v));  // silu
  }
  __syncthreads();
  float acc = bias[n];
  for (int k = 0; k < D; ++k) acc += st[k] * W[(size_t)k * N3 + n];
  E[(size_t)b * N3 + n] = acc;
}

// ---------------------------------------------------------------------------
// adaLN: layernorm(x) * (1 + scale) + shift -> bf16    (block per row)
// E layout per batch: [shift(D) | scale(D) | gate(D)]
// ---------------------------------------------------------------------------
__global__ __launch_bounds__(256) void adaln_kernel(
    const float* __restrict__ Xin, const float* __restrict__ E,
    bf16* __restrict__ Xout, int S, int D) {
  int row = blockIdx.x;  // b*S + s
  int b = row / S;
  const float* x = Xin + (size_t)row * D;
  int lane = threadIdx.x & 31;
  int w = threadIdx.x >> 5;
  __shared__ float red[16];
  float sum = 0.f, sq = 0.f;
  for (int i = threadIdx.x; i < D; i += 256) {
    float v = x[i];
    sum += v;
    sq += v * v;
  }
#pragma unroll
  for (int off = 1; off < 32; off <<= 1) {
    sum += __shfl_xor(sum, off, 32);
    sq += __shfl_xor(sq, off, 32);
  }
  if (lane == 0) {
    red[w * 2] = sum;
    red[w * 2 + 1] = sq;
  }
  __syncthreads();
  sum = 0.f;
  sq = 0.f;
#pragma unroll
  for (int i = 0; i < 8; ++i) {
    sum += red[i * 2];
    sq += red[i * 2 + 1];
  }
  float mu = sum / D;
  float var = sq / D - mu * mu;
  float inv = rsqrtf(var + 1e-6f);
  const float* sh = E + (size_t)b * 3 * D;
  const float* sc = sh + D;
  for (int i = threadIdx.x; i < D; i += 256) {
    float xn = (x[i] - mu) * inv;
    Xout[(size_t)row * D + i] = (bf16)(xn * (1.0f + sc[i]) + sh[i]);
  }
}

// ---------------------------------------------------------------------------
// WMMA GEMM: C(MxN) = A(MxK,bf16) * Wt(NxK,bf16)^T + bias
// Block tile 256x128, K-step 32, double-buffered LDS with async copies.
// 8 waves (4 along M x 2 along N), each wave 64x64 = 4x4 wmma accumulators.
// mode 0: f32 out      mode 1: bf16 out     mode 2: bf16(gelu(out))
// mode 3: f32 out = base + gate[n] * out    (fused residual)
// ---------------------------------------------------------------------------
#define BM 256
#define BN 128
#define BKK 32

__global__ __launch_bounds__(256) void gemm_bf16(
    const bf16* __restrict__ A, const bf16* __restrict__ Wt,
    const float* __restrict__ bias, void* __restrict__ Cout, int M, int N,
    int K, int ldc, int mode, const float* __restrict__ base,
    const float* __restrict__ gate) {
  __shared__ __align__(32) bf16 sA[2][BM * BKK];  // 2 x 16 KB
  __shared__ __align__(32) bf16 sB[2][BN * BKK];  // 2 x  8 KB
  int tid = threadIdx.x;
  int lane = tid & 31;
  int w = tid >> 5;
  int wm = w & 3;   // 4 waves along M (64 rows each)
  int wn = w >> 2;  // 2 waves along N (64 cols each)
  int half = lane >> 4;
  int l16 = lane & 15;
  int bm0 = blockIdx.x * BM;
  int bn0 = blockIdx.y * BN;

  f32x8 acc[4][4];
#pragma unroll
  for (int i = 0; i < 4; ++i)
#pragma unroll
    for (int j = 0; j < 4; ++j) acc[i][j] = {};

  // per-tile copies: sA = 1024 16B chunks (4/thread), sB = 512 (2/thread)
  auto issue_tile = [&](int buf, int k0) {
#pragma unroll
    for (int c = 0; c < 4; ++c) {
      int cc = tid + c * 256;
      int row = cc >> 2;
      int kc = (cc & 3) * 8;
      TILE_COPY(&sA[buf][row * BKK + kc],
                &A[(size_t)(bm0 + row) * K + k0 + kc]);
    }
#pragma unroll
    for (int c = 0; c < 2; ++c) {
      int cc = tid + c * 256;
      int row = cc >> 2;
      int kc = (cc & 3) * 8;
      TILE_COPY(&sB[buf][row * BKK + kc],
                &Wt[(size_t)(bn0 + row) * K + k0 + kc]);
    }
  };

  int nst = K / BKK;
  issue_tile(0, 0);
  for (int i = 0; i < nst; ++i) {
    int cur = i & 1;
    if (i + 1 < nst) {
      issue_tile(cur ^ 1, (i + 1) * BKK);
      WAIT_ASYNC(6);  // current tile's 6 async ops have retired
    } else {
      WAIT_ASYNC(0);
    }
    __syncthreads();

    bf16x16 bfm[4];
#pragma unroll
    for (int ni = 0; ni < 4; ++ni) {
      int n = wn * 64 + ni * 16 + l16;
      bfm[ni] = cat8(*(const bf16x8*)&sB[cur][n * BKK + half * 8],
                     *(const bf16x8*)&sB[cur][n * BKK + 16 + half * 8]);
    }
#pragma unroll
    for (int mi = 0; mi < 4; ++mi) {
      int m = wm * 64 + mi * 16 + l16;
      bf16x8 lo = *(const bf16x8*)&sA[cur][m * BKK + half * 8];
      bf16x8 hi = *(const bf16x8*)&sA[cur][m * BKK + 16 + half * 8];
      bf16x16 af = cat8(lo, hi);
#pragma unroll
      for (int ni = 0; ni < 4; ++ni)
        acc[mi][ni] = wmma_bf16(af, bfm[ni], acc[mi][ni]);
    }
    __syncthreads();
  }

#pragma unroll
  for (int mi = 0; mi < 4; ++mi) {
#pragma unroll
    for (int ni = 0; ni < 4; ++ni) {
      int n = bn0 + wn * 64 + ni * 16 + l16;
      float bv = bias[n];
#pragma unroll
      for (int v = 0; v < 8; ++v) {
        int m = bm0 + wm * 64 + mi * 16 + v + half * 8;
        size_t idx = (size_t)m * ldc + n;
        float x = acc[mi][ni][v] + bv;
        if (mode == 0) {
          ((float*)Cout)[idx] = x;
        } else if (mode == 1) {
          ((bf16*)Cout)[idx] = (bf16)x;
        } else if (mode == 2) {
          ((bf16*)Cout)[idx] = (bf16)gelu_tanh(x);
        } else {
          ((float*)Cout)[idx] = base[idx] + gate[n] * x;
        }
      }
    }
  }
}

// ---------------------------------------------------------------------------
// RMS-norm (per head) + RoPE, in place on bf16 buffer. One wave per (b,s,h).
// ---------------------------------------------------------------------------
__global__ __launch_bounds__(256) void qk_post(
    bf16* __restrict__ X, const float* __restrict__ rcos,
    const float* __restrict__ rsin, const float* __restrict__ rms_ctx,
    const float* __restrict__ rms_noi, int S, int D, int H, int HD, int CTX) {
  int gw = (blockIdx.x * 256 + threadIdx.x) >> 5;
  int lane = threadIdx.x & 31;
  int h = gw % H;
  int s = (gw / H) % S;
  int b = gw / (H * S);
  bf16* p = X + ((size_t)b * S + s) * D + h * HD + lane * 4;
  float x0 = (float)p[0], x1 = (float)p[1], x2 = (float)p[2], x3 = (float)p[3];
  float ss = x0 * x0 + x1 * x1 + x2 * x2 + x3 * x3;
#pragma unroll
  for (int off = 1; off < 32; off <<= 1) ss += __shfl_xor(ss, off, 32);
  float r = rsqrtf(ss / (float)HD + 1e-6f);
  const float* rw = (s < CTX) ? rms_ctx : rms_noi;
  int hd = lane * 4;
  x0 *= r * rw[hd];
  x1 *= r * rw[hd + 1];
  x2 *= r * rw[hd + 2];
  x3 *= r * rw[hd + 3];
  const float* cs = rcos + (size_t)s * HD + hd;
  const float* sn = rsin + (size_t)s * HD + hd;
  float o0 = x0 * cs[0] - x1 * sn[0];
  float o1 = x1 * cs[1] + x0 * sn[1];
  float o2 = x2 * cs[2] - x3 * sn[2];
  float o3 = x3 * cs[3] + x2 * sn[3];
  p[0] = (bf16)o0;
  p[1] = (bf16)o1;
  p[2] = (bf16)o2;
  p[3] = (bf16)o3;
}

// ---------------------------------------------------------------------------
// Flash attention (WMMA, online softmax). Block = 128 q rows of one (b,h).
// 8 waves x 16 q rows. Streams 32-key K/V blocks through LDS.
// ---------------------------------------------------------------------------
__global__ __launch_bounds__(256) void flash_attn(
    const bf16* __restrict__ Q, const bf16* __restrict__ Kb,
    const bf16* __restrict__ V, bf16* __restrict__ O, int S, int D, int H,
    int HD, float scale) {
  __shared__ __align__(32) bf16 sK[32 * 128];   // [key][hd]
  __shared__ __align__(32) bf16 sVt[128 * 32];  // [hd][key]
  __shared__ __align__(32) bf16 sP[8][16 * 32]; // per-wave [m][key]
  int tid = threadIdx.x;
  int lane = tid & 31;
  int w = tid >> 5;
  int half = lane >> 4;
  int l16 = lane & 15;
  int b = blockIdx.z;
  int h = blockIdx.y;
  int qrow0 = blockIdx.x * 128 + w * 16;

  // Q fragments: A-matrix layout, 4 chunks of K=32 along HD
  bf16x16 aq[4];
#pragma unroll
  for (int c = 0; c < 4; ++c) {
    const bf16* qp =
        Q + ((size_t)b * S + qrow0 + l16) * D + h * HD + c * 32 + half * 8;
    bf16x8 lo = *(const bf16x8*)qp;
    bf16x8 hi = *(const bf16x8*)(qp + 16);
    aq[c] = cat8(lo, hi);
  }

  float mstat[8], lstat[8];
  f32x8 accO[8];
#pragma unroll
  for (int v = 0; v < 8; ++v) {
    mstat[v] = -1e30f;
    lstat[v] = 0.f;
  }
#pragma unroll
  for (int j = 0; j < 8; ++j) accO[j] = {};

  for (int kb = 0; kb < S; kb += 32) {
    // stage K block [32][128]
#pragma unroll
    for (int c = tid; c < 32 * 128 / 8; c += 256) {
      int r = c >> 4;
      int cc = (c & 15) * 8;
      *(bf16x8*)&sK[r * 128 + cc] =
          *(const bf16x8*)&Kb[((size_t)b * S + kb + r) * D + h * HD + cc];
    }
    // stage V transposed [128][32]
    {
      int e0 = tid * 16;  // 4096 elements / 256 threads = 16 each
      int key = e0 >> 7;
      int hd0 = e0 & 127;
      const bf16* vp = &V[((size_t)b * S + kb + key) * D + h * HD + hd0];
#pragma unroll
      for (int e = 0; e < 16; ++e) sVt[(hd0 + e) * 32 + key] = vp[e];
    }
    __syncthreads();

    // scores: two 16-key sub-tiles
    f32x8 sc[2];
#pragma unroll
    for (int t = 0; t < 2; ++t) {
      f32x8 c0 = {};
#pragma unroll
      for (int ch = 0; ch < 4; ++ch) {
        bf16x16 bk = cat8(
            *(const bf16x8*)&sK[(t * 16 + l16) * 128 + ch * 32 + half * 8],
            *(const bf16x8*)&sK[(t * 16 + l16) * 128 + ch * 32 + 16 + half * 8]);
        c0 = wmma_bf16(aq[ch], bk, c0);
      }
      sc[t] = c0;
    }

    // online softmax (row = v + half*8, cols across l16 lanes)
    float alpha[8];
#pragma unroll
    for (int v = 0; v < 8; ++v) {
      float s0 = sc[0][v] * scale;
      float s1 = sc[1][v] * scale;
      float rm = fmaxf(s0, s1);
#pragma unroll
      for (int off = 1; off < 16; off <<= 1)
        rm = fmaxf(rm, __shfl_xor(rm, off, 32));
      float mnew = fmaxf(mstat[v], rm);
      alpha[v] = __expf(mstat[v] - mnew);
      float p0 = __expf(s0 - mnew);
      float p1 = __expf(s1 - mnew);
      float rs = p0 + p1;
#pragma unroll
      for (int off = 1; off < 16; off <<= 1) rs += __shfl_xor(rs, off, 32);
      lstat[v] = lstat[v] * alpha[v] + rs;
      mstat[v] = mnew;
      sc[0][v] = p0;
      sc[1][v] = p1;
    }
#pragma unroll
    for (int j = 0; j < 8; ++j)
#pragma unroll
      for (int v = 0; v < 8; ++v) accO[j][v] *= alpha[v];

    // stage P through LDS to convert C-layout -> A-layout
    bf16* pw = &sP[w][0];
#pragma unroll
    for (int t = 0; t < 2; ++t)
#pragma unroll
      for (int v = 0; v < 8; ++v)
        pw[(v + half * 8) * 32 + t * 16 + l16] = (bf16)sc[t][v];
    bf16x8 plo = *(const bf16x8*)&pw[l16 * 32 + half * 8];
    bf16x8 phi = *(const bf16x8*)&pw[l16 * 32 + 16 + half * 8];
    bf16x16 ap = cat8(plo, phi);

    // O += P @ V
#pragma unroll
    for (int j = 0; j < 8; ++j) {
      bf16x16 bv = cat8(*(const bf16x8*)&sVt[(j * 16 + l16) * 32 + half * 8],
                        *(const bf16x8*)&sVt[(j * 16 + l16) * 32 + 16 + half * 8]);
      accO[j] = wmma_bf16(ap, bv, accO[j]);
    }
    __syncthreads();
  }

  // normalize + store
#pragma unroll
  for (int j = 0; j < 8; ++j) {
#pragma unroll
    for (int v = 0; v < 8; ++v) {
      int m = qrow0 + v + half * 8;
      int n = h * HD + j * 16 + l16;
      O[((size_t)b * S + m) * D + n] = (bf16)(accO[j][v] / lstat[v]);
    }
  }
}

// ---------------------------------------------------------------------------
// Host-side orchestration
// ---------------------------------------------------------------------------
extern "C" void kernel_launch(void* const* d_in, const int* in_sizes, int n_in,
                              void* d_out, int out_size, void* d_ws,
                              size_t ws_size, hipStream_t stream) {
  (void)in_sizes;
  (void)n_in;
  (void)out_size;
  (void)ws_size;
  constexpr int B = 2, S = 2048, D = 2048, H = 16, HD = 128, CTX = 1792;
  constexpr int NOI = S - CTX;
  constexpr int D3 = 3 * D, D4 = 4 * D;

  const float* hidden = (const float*)d_in[0];
  const float* temb = (const float*)d_in[1];
  const float* rcos = (const float*)d_in[2];
  const float* rsin = (const float*)d_in[3];
  const float* n1w = (const float*)d_in[4];
  const float* n1b = (const float*)d_in[5];
  const float* q_w = (const float*)d_in[6];
  const float* q_b = (const float*)d_in[7];
  const float* k_w = (const float*)d_in[8];
  const float* k_b = (const float*)d_in[9];
  const float* v_w = (const float*)d_in[10];
  const float* v_b = (const float*)d_in[11];
  const float* aq_w = (const float*)d_in[12];
  const float* aq_b = (const float*)d_in[13];
  const float* ak_w = (const float*)d_in[14];
  const float* ak_b = (const float*)d_in[15];
  const float* av_w = (const float*)d_in[16];
  const float* av_b = (const float*)d_in[17];
  const float* rms_q = (const float*)d_in[18];
  const float* rms_k = (const float*)d_in[19];
  const float* rms_aq = (const float*)d_in[20];
  const float* rms_ak = (const float*)d_in[21];
  const float* out_w = (const float*)d_in[22];
  const float* out_b = (const float*)d_in[23];
  const float* aout_w = (const float*)d_in[24];
  const float* aout_b = (const float*)d_in[25];
  const float* n2w = (const float*)d_in[26];
  const float* n2b = (const float*)d_in[27];
  const float* w1 = (const float*)d_in[28];
  const float* b1 = (const float*)d_in[29];
  const float* w2 = (const float*)d_in[30];
  const float* b2 = (const float*)d_in[31];
  float* out = (float*)d_out;

  char* ws = (char*)d_ws;
  size_t off = 0;
  auto alloc = [&](size_t bytes) -> char* {
    char* p = ws + off;
    off = (off + bytes + 255) & ~(size_t)255;
    return p;
  };
  bf16* wt_q = (bf16*)alloc((size_t)D * D * 2);
  bf16* wt_k = (bf16*)alloc((size_t)D * D * 2);
  bf16* wt_v = (bf16*)alloc((size_t)D * D * 2);
  bf16* wt_aq = (bf16*)alloc((size_t)D * D * 2);
  bf16* wt_ak = (bf16*)alloc((size_t)D * D * 2);
  bf16* wt_av = (bf16*)alloc((size_t)D * D * 2);
  bf16* wt_o = (bf16*)alloc((size_t)D * D * 2);
  bf16* wt_ao = (bf16*)alloc((size_t)D * D * 2);
  bf16* wt_1 = (bf16*)alloc((size_t)D * D4 * 2);
  bf16* wt_2 = (bf16*)alloc((size_t)D4 * D * 2);
  float* e1 = (float*)alloc((size_t)B * D3 * 4);
  float* e2 = (float*)alloc((size_t)B * D3 * 4);
  bf16* xb = (bf16*)alloc((size_t)B * S * D * 2);
  bf16* qb = (bf16*)alloc((size_t)B * S * D * 2);
  bf16* kb = (bf16*)alloc((size_t)B * S * D * 2);
  bf16* vb = (bf16*)alloc((size_t)B * S * D * 2);
  bf16* ob = (bf16*)alloc((size_t)B * S * D * 2);
  float* hbuf = (float*)alloc((size_t)B * S * D * 4);
  bf16* x2b = (bf16*)alloc((size_t)B * S * D * 2);
  bf16* m1b = (bf16*)alloc((size_t)B * S * D4 * 2);

  auto conv = [&](const float* W, bf16* Wt, int kshift, int N) {
    size_t tot = ((size_t)1 << kshift) * N;
    conv_wT<<<(unsigned)((tot + 255) / 256), 256, 0, stream>>>(W, Wt, kshift, N);
  };
  conv(q_w, wt_q, 11, D);
  conv(k_w, wt_k, 11, D);
  conv(v_w, wt_v, 11, D);
  conv(aq_w, wt_aq, 11, D);
  conv(ak_w, wt_ak, 11, D);
  conv(av_w, wt_av, 11, D);
  conv(out_w, wt_o, 11, D);
  conv(aout_w, wt_ao, 11, D);
  conv(w1, wt_1, 11, D4);
  conv(w2, wt_2, 13, D);

  temb_gemm<<<dim3(D3 / 256, B), 256, 0, stream>>>(temb, n1w, n1b, e1, D, D3);
  temb_gemm<<<dim3(D3 / 256, B), 256, 0, stream>>>(temb, n2w, n2b, e2, D, D3);

  adaln_kernel<<<B * S, 256, 0, stream>>>(hidden, e1, xb, S, D);

  auto gemm = [&](const bf16* A, const bf16* Wt, const float* bias, void* C,
                  int M, int N, int K, int ldc, int mode, const float* base,
                  const float* gate) {
    gemm_bf16<<<dim3(M / BM, N / BN), 256, 0, stream>>>(A, Wt, bias, C, M, N,
                                                        K, ldc, mode, base,
                                                        gate);
  };

  // QKV projections (ctx rows use a*_w, noise rows use *_w)
  for (int b = 0; b < B; ++b) {
    size_t rc = (size_t)b * S;        // ctx row start
    size_t rn = (size_t)b * S + CTX;  // noise row start
    gemm(xb + rc * D, wt_aq, aq_b, qb + rc * D, CTX, D, D, D, 1, 0, 0);
    gemm(xb + rc * D, wt_ak, ak_b, kb + rc * D, CTX, D, D, D, 1, 0, 0);
    gemm(xb + rc * D, wt_av, av_b, vb + rc * D, CTX, D, D, D, 1, 0, 0);
    gemm(xb + rn * D, wt_q, q_b, qb + rn * D, NOI, D, D, D, 1, 0, 0);
    gemm(xb + rn * D, wt_k, k_b, kb + rn * D, NOI, D, D, D, 1, 0, 0);
    gemm(xb + rn * D, wt_v, v_b, vb + rn * D, NOI, D, D, D, 1, 0, 0);
  }

  // RMS + RoPE on q and k (ctx / noise rms weights)
  qk_post<<<B * S * H / 8, 256, 0, stream>>>(qb, rcos, rsin, rms_aq, rms_q, S,
                                             D, H, HD, CTX);
  qk_post<<<B * S * H / 8, 256, 0, stream>>>(kb, rcos, rsin, rms_ak, rms_k, S,
                                             D, H, HD, CTX);

  // Attention
  flash_attn<<<dim3(S / 128, H, B), 256, 0, stream>>>(
      qb, kb, vb, ob, S, D, H, HD, 0.08838834764831845f);

  // Output projection fused with residual: h = hidden + gate1 * (o@W + b)
  for (int b = 0; b < B; ++b) {
    size_t rc = (size_t)b * S;
    size_t rn = (size_t)b * S + CTX;
    const float* g1 = e1 + (size_t)b * D3 + 2 * D;
    gemm(ob + rc * D, wt_ao, aout_b, hbuf + rc * D, CTX, D, D, D, 3,
         hidden + rc * D, g1);
    gemm(ob + rn * D, wt_o, out_b, hbuf + rn * D, NOI, D, D, D, 3,
         hidden + rn * D, g1);
  }

  // adaLN 2
  adaln_kernel<<<B * S, 256, 0, stream>>>(hbuf, e2, x2b, S, D);

  // MLP up (gelu fused)
  gemm(x2b, wt_1, b1, m1b, B * S, D4, D, D4, 2, 0, 0);

  // MLP down fused with final residual: out = h + gate2 * (m1@W2 + b2)
  for (int b = 0; b < B; ++b) {
    size_t r0 = (size_t)b * S;
    const float* g2 = e2 + (size_t)b * D3 + 2 * D;
    gemm(m1b + r0 * D4, wt_2, b2, out + r0 * D, S, D, D4, D, 3, hbuf + r0 * D,
         g2);
  }
}
